// MoE_38242388803777
// MI455X (gfx1250) — matrix operand order repaired and checked by reading, not compiled, not run on hardware
//
#include <hip/hip_runtime.h>
#include <hip/hip_bf16.h>
#include <math.h>

// ---------------- problem constants (match reference) ----------------
constexpr int TB    = 2048;  // batch
constexpr int DIM   = 1024;
constexpr int INTER = 2048;
constexpr int NE    = 8;
constexpr int LANG  = 768;
constexpr int SH    = 4096;

// ---------------- WMMA tile config ----------------
constexpr int BM = 128, BN = 64, BK = 32;
constexpr int LDA = BK + 8;           // padded LDS row stride (bf16 elems) for A
constexpr int LDB = BK + 8;           // padded LDS row stride (bf16 elems) for B^T
constexpr int ABYTES = BM * LDA * 2;  // one A buffer, bytes

typedef __attribute__((ext_vector_type(16))) __bf16 v16bf;
typedef __attribute__((ext_vector_type(8)))  float  v8f;

__device__ __forceinline__ unsigned short f2bf(float f) {
  unsigned int u = __float_as_uint(f);
  u += 0x7fffu + ((u >> 16) & 1u);    // round-to-nearest-even
  return (unsigned short)(u >> 16);
}

union FragU { v16bf v; uint4 q[2]; };

__device__ __forceinline__ v8f wmma_bf16(v16bf a, v16bf b, v8f c) {
  return __builtin_amdgcn_wmma_f32_16x16x32_bf16(false, a, false, b,
                                                 (short)0, c, false, false);
}

// A fragment (16x32 bf16): lane<16: e0-7=K0-7, e8-15=K16-23 of row(lane&15)
//                          lane>=16: e0-7=K8-15, e8-15=K24-31
__device__ __forceinline__ v16bf load_a_frag(const unsigned short* As, int mrow, int lane) {
  const unsigned short* base = As + mrow * LDA + ((lane & 16) ? 8 : 0);
  FragU u;
  u.q[0] = *(const uint4*)(base);
  u.q[1] = *(const uint4*)(base + 16);
  return u.v;
}

// B fragment from transposed tile Bt[n][k]: lane<16: e0-7=K0-7, e8-15=K16-23 of col(lane&15); lane>=16: e0-7=K8-15, e8-15=K24-31
__device__ __forceinline__ v16bf load_b_frag(const unsigned short* Bt, int ncol, int lane) {
  const unsigned short* base = Bt + ncol * LDB + ((lane & 16) ? 8 : 0);
  FragU u;
  u.q[0] = *(const uint4*)(base);
  u.q[1] = *(const uint4*)(base + 16);
  return u.v;
}

// ---- CDNA5 async global->LDS copy (16B per lane), ASYNCcnt tracked ----
__device__ __forceinline__ void async_copy_b128(unsigned lds_off, const unsigned short* gptr) {
  asm volatile("global_load_async_to_lds_b128 %0, %1, off"
               :: "v"(lds_off), "v"(gptr) : "memory");
}
__device__ __forceinline__ void wait_async() {
  asm volatile("s_wait_asynccnt 0x0" ::: "memory");
}

// pack fp32 rows k2,k2+1 (4 cols) into u32 pairs of the transposed bf16 tile
__device__ __forceinline__ void store_pack4(unsigned short* bt, int gcol, int k2,
                                            float4 fa, float4 fb) {
  unsigned* p = (unsigned*)bt;
  int kk = k2 >> 1;
  p[(gcol + 0) * (LDB / 2) + kk] = (unsigned)f2bf(fa.x) | ((unsigned)f2bf(fb.x) << 16);
  p[(gcol + 1) * (LDB / 2) + kk] = (unsigned)f2bf(fa.y) | ((unsigned)f2bf(fb.y) << 16);
  p[(gcol + 2) * (LDB / 2) + kk] = (unsigned)f2bf(fa.z) | ((unsigned)f2bf(fb.z) << 16);
  p[(gcol + 3) * (LDB / 2) + kk] = (unsigned)f2bf(fa.w) | ((unsigned)f2bf(fb.w) << 16);
}

// ---------------- gate: top-1 routing (one wave32 per token) ----------------
__global__ __launch_bounds__(256) void gate_kernel(const float* __restrict__ lt,
                                                   const float* __restrict__ re,
                                                   int* __restrict__ counts,
                                                   int* __restrict__ list) {
  int wave = threadIdx.x >> 5, lane = threadIdx.x & 31;
  int b = blockIdx.x * 8 + wave;
  const float* t = lt + (long)b * LANG;
  float best = -3.4e38f; int bestE = 0;
  for (int e = 0; e < NE; ++e) {
    const float* r = re + (long)e * LANG;
    float d = 0.f, n = 0.f;
    for (int i = lane; i < LANG; i += 32) { float rv = r[i]; d += t[i] * rv; n += rv * rv; }
#pragma unroll
    for (int m = 16; m; m >>= 1) { d += __shfl_xor(d, m, 32); n += __shfl_xor(n, m, 32); }
    float sim = d * rsqrtf(fmaxf(n, 1e-24f));   // token norm cancels in argmax
    if (sim > best) { best = sim; bestE = e; }  // strict > == jnp.argmax tie rule
  }
  if (lane == 0) {
    int pos = atomicAdd(&counts[bestE], 1);
    list[bestE * TB + pos] = b;
  }
}

// ---------------- fp32 -> bf16 activation convert ----------------
__global__ void cvt_kernel(const float* __restrict__ x, unsigned short* __restrict__ xbf, int n) {
  int i = blockIdx.x * blockDim.x + threadIdx.x;
  if (i < n) xbf[i] = f2bf(x[i]);
}

// ---------------- unified pipelined WMMA GEMM ----------------
// MODE 0: S1  (direct A, 1 B, silu -> bf16 store)
// MODE 1: S2  (direct A, 1 B, +bias -> f32 store, overwrites d_out)
// MODE 2: FFN1 (gathered A, dual B, swiglu -> bf16 scatter by token)
// MODE 3: FFN2 (gathered A, 1 B, scale+bias -> f32 accumulate scatter)
struct GArgs {
  const unsigned short* A;
  const float* B0;
  const float* B1;
  const float* bias0;
  const float* bias1;
  const float* ew;
  const int* counts;
  const int* list;
  unsigned short* obf;
  float* of32;
  int K;
  int Astride;
  int Nstride;
};

template <int MODE>
__global__ __launch_bounds__(256) void gemm_kernel(GArgs g) {
  constexpr bool GATHER = (MODE >= 2);
  constexpr bool DUAL   = (MODE == 2);
  constexpr int  NB     = DUAL ? 2 : 1;

  int e = GATHER ? blockIdx.z : 0;
  int cnt = TB;
  const int* el = nullptr;
  if constexpr (GATHER) { cnt = g.counts[e]; el = g.list + e * TB; }
  int m0 = blockIdx.y * BM;
  if (m0 >= cnt) return;                         // block-uniform: EXEC stays all-1s for WMMA
  int n0 = blockIdx.x * BN;

  const float* B0 = g.B0;
  const float* B1 = g.B1;
  if constexpr (GATHER) {
    long wsz = (long)g.K * g.Nstride;
    B0 += (long)e * wsz;
    if constexpr (DUAL) B1 += (long)e * wsz;
  }
  float we = 0.f;
  if constexpr (MODE == 3) we = g.ew[e];

  __shared__ __align__(16) unsigned short As[2][BM * LDA];
  __shared__ __align__(16) unsigned short Bs[2][NB * BN * LDB];

  int tid = threadIdx.x, lane = tid & 31, wave = tid >> 5;
  int wm = (wave >> 1) * 32, wn = (wave & 1) * 32;

  // ---- K-invariant staging addresses (2 A slots + 1 B slot per thread) ----
  const unsigned short* aSrc[2];
  unsigned aDst[2];
#pragma unroll
  for (int it = 0; it < 2; ++it) {
    int slot = tid + it * 256;                   // BM*BK/8 = 512 slots of 8 bf16
    int r = slot >> 2, c = (slot & 3) * 8;
    long grow;
    if constexpr (GATHER) grow = el[min(m0 + r, cnt - 1)];  // clamp: OOB rows never stored
    else                  grow = m0 + r;
    aSrc[it] = g.A + grow * (long)g.Astride + c;
    aDst[it] = (unsigned)(size_t)(&As[0][r * LDA + c]);
  }
  int k2 = (tid >> 4) * 2, gcol = (tid & 15) * 4;
  const float* bSrc0 = B0 + (long)k2 * g.Nstride + n0 + gcol;
  const float* bSrc1 = DUAL ? (B1 + (long)k2 * g.Nstride + n0 + gcol) : nullptr;

  v8f acc[2][2]  = {};
  v8f acc3[2][2] = {};

  auto stage = [&](int kblk, int buf) {
    // A tile: async DMA, no VGPR round-trip (src already bf16)
#pragma unroll
    for (int it = 0; it < 2; ++it)
      async_copy_b128(aDst[it] + buf * ABYTES, aSrc[it] + kblk);
    // B tile(s): fp32 -> packed bf16 pairs -> transposed LDS (b32 stores)
    const float* p0 = bSrc0 + (long)kblk * g.Nstride;
    float4 f0 = *(const float4*)p0;
    float4 f1 = *(const float4*)(p0 + g.Nstride);
    store_pack4(&Bs[buf][0], gcol, k2, f0, f1);
    if constexpr (DUAL) {
      const float* p1 = bSrc1 + (long)kblk * g.Nstride;
      float4 h0 = *(const float4*)p1;
      float4 h1 = *(const float4*)(p1 + g.Nstride);
      store_pack4(&Bs[buf][BN * LDB], gcol, k2, h0, h1);
    }
  };

  auto compute = [&](int buf) {
    const unsigned short* as = &As[buf][0];
    const unsigned short* bt = &Bs[buf][0];
    int mr = lane & 15;
    v16bf a0 = load_a_frag(as, wm + mr, lane);
    v16bf a1 = load_a_frag(as, wm + 16 + mr, lane);
    v16bf p0 = load_b_frag(bt, wn + mr, lane);
    v16bf p1 = load_b_frag(bt, wn + 16 + mr, lane);
    acc[0][0] = wmma_bf16(a0, p0, acc[0][0]);
    acc[0][1] = wmma_bf16(a0, p1, acc[0][1]);
    acc[1][0] = wmma_bf16(a1, p0, acc[1][0]);
    acc[1][1] = wmma_bf16(a1, p1, acc[1][1]);
    if constexpr (DUAL) {
      const unsigned short* bt3 = bt + BN * LDB;
      v16bf q0 = load_b_frag(bt3, wn + mr, lane);
      v16bf q1 = load_b_frag(bt3, wn + 16 + mr, lane);
      acc3[0][0] = wmma_bf16(a0, q0, acc3[0][0]);
      acc3[0][1] = wmma_bf16(a0, q1, acc3[0][1]);
      acc3[1][0] = wmma_bf16(a1, q0, acc3[1][0]);
      acc3[1][1] = wmma_bf16(a1, q1, acc3[1][1]);
    }
  };

  // ---- double-buffered pipeline: prefetch k+1 while WMMAs consume k ----
  stage(0, 0);
  wait_async();
  __syncthreads();
  int K = g.K;
  for (int k0 = 0; k0 < K; k0 += BK) {
    int buf = (k0 / BK) & 1;
    if (k0 + BK < K) stage(k0 + BK, buf ^ 1);
    compute(buf);
    wait_async();
    __syncthreads();
  }

  // ---- epilogue ----
  int colB = lane & 15, rowAdd = (lane & 16) ? 8 : 0;
#pragma unroll
  for (int mi = 0; mi < 2; ++mi)
#pragma unroll
    for (int ni = 0; ni < 2; ++ni)
#pragma unroll
      for (int r = 0; r < 8; ++r) {
        int lm = m0 + wm + mi * 16 + r + rowAdd;
        int gn = n0 + wn + ni * 16 + colB;
        float v = acc[mi][ni][r];
        if constexpr (MODE == 0) {
          v += g.bias0[gn];
          float sv = v / (1.0f + __expf(-v));
          g.obf[(long)lm * g.Nstride + gn] = f2bf(sv);
        } else if constexpr (MODE == 1) {
          g.of32[(long)lm * g.Nstride + gn] = v + g.bias0[gn];
        } else if constexpr (MODE == 2) {
          if (lm < cnt) {
            int tok = el[lm];
            float h1 = v + g.bias0[e * g.Nstride + gn];
            float h3 = acc3[mi][ni][r] + g.bias1[e * g.Nstride + gn];
            float sv = (h1 / (1.0f + __expf(-h1))) * h3;
            g.obf[(long)tok * g.Nstride + gn] = f2bf(sv);
          }
        } else {
          if (lm < cnt) {
            int tok = el[lm];
            g.of32[(long)tok * g.Nstride + gn] += (v + g.bias0[e * g.Nstride + gn]) * we;
          }
        }
      }
}

extern "C" void kernel_launch(void* const* d_in, const int* in_sizes, int n_in,
                              void* d_out, int out_size, void* d_ws, size_t ws_size,
                              hipStream_t stream) {
  (void)in_sizes; (void)n_in; (void)out_size; (void)ws_size;
  const float* x   = (const float*)d_in[0];
  const float* lt  = (const float*)d_in[1];
  const float* re  = (const float*)d_in[2];
  const float* ew  = (const float*)d_in[3];
  const float* W1  = (const float*)d_in[4];
  const float* b1  = (const float*)d_in[5];
  const float* W2  = (const float*)d_in[6];
  const float* b2  = (const float*)d_in[7];
  const float* W3  = (const float*)d_in[8];
  const float* b3  = (const float*)d_in[9];
  const float* Ws1 = (const float*)d_in[10];
  const float* bs1 = (const float*)d_in[11];
  const float* Ws2 = (const float*)d_in[12];
  const float* bs2 = (const float*)d_in[13];
  float* out = (float*)d_out;

  // workspace layout (~28.2 MB total)
  char* ws = (char*)d_ws;
  int* counts = (int*)ws;                                   // 32 B
  int* list   = (int*)(ws + 256);                           // 64 KB
  unsigned short* xbf  = (unsigned short*)(ws + (1 << 17)); // 4 MB
  unsigned short* Hbuf = xbf + (size_t)TB * DIM;            // 8 MB
  unsigned short* Sbuf = Hbuf + (size_t)TB * INTER;         // 16 MB

  hipMemsetAsync(counts, 0, NE * sizeof(int), stream);
  gate_kernel<<<TB / 8, 256, 0, stream>>>(lt, re, counts, list);
  cvt_kernel<<<(TB * DIM + 255) / 256, 256, 0, stream>>>(x, xbf, TB * DIM);

  GArgs a1{};  // FFN1
  a1.A = xbf; a1.B0 = W1; a1.B1 = W3; a1.bias0 = b1; a1.bias1 = b3;
  a1.counts = counts; a1.list = list; a1.obf = Hbuf;
  a1.K = DIM; a1.Astride = DIM; a1.Nstride = INTER;
  gemm_kernel<2><<<dim3(INTER / BN, TB / BM, NE), 256, 0, stream>>>(a1);

  GArgs s1{};  // shared stage 1
  s1.A = xbf; s1.B0 = Ws1; s1.bias0 = bs1; s1.obf = Sbuf;
  s1.K = DIM; s1.Astride = DIM; s1.Nstride = SH;
  gemm_kernel<0><<<dim3(SH / BN, TB / BM, 1), 256, 0, stream>>>(s1);

  GArgs s2{};  // shared stage 2 (fully overwrites d_out)
  s2.A = Sbuf; s2.B0 = Ws2; s2.bias0 = bs2; s2.of32 = out;
  s2.K = SH; s2.Astride = SH; s2.Nstride = DIM;
  gemm_kernel<1><<<dim3(DIM / BN, TB / BM, 1), 256, 0, stream>>>(s2);

  GArgs a2{};  // FFN2 (accumulates after S2)
  a2.A = Hbuf; a2.B0 = W2; a2.bias0 = b2; a2.ew = ew;
  a2.counts = counts; a2.list = list; a2.of32 = out;
  a2.K = INTER; a2.Astride = INTER; a2.Nstride = DIM;
  gemm_kernel<3><<<dim3(DIM / BN, TB / BM, NE), 256, 0, stream>>>(a2);
}
